// MLPComponentsFullRank_15848429322668
// MI455X (gfx1250) — compile-verified, matrix operand order checked by reading, not run on hardware
//
#include <hip/hip_runtime.h>

// ---------------------------------------------------------------------------
// MoE masked 2-layer MLP for MI455X (gfx1250, wave32, WMMA).
// bf16x3 split-precision GEMM on v_wmma_f32_16x16x32_bf16.
// Fast path: operands pre-split to bf16 hi/lo in d_ws, tiles staged with
// global_load_async_to_lds_b128 + s_wait_asynccnt, double-buffered LDS.
// Fallback (small d_ws): in-kernel f32->bf16 split staging.
// ---------------------------------------------------------------------------

typedef __attribute__((ext_vector_type(16))) __bf16 v16bf;
typedef __attribute__((ext_vector_type(8)))  __bf16 v8bf;
typedef __attribute__((ext_vector_type(8)))  float  v8f;

#define DEV __device__ __forceinline__

constexpr int BATCH = 2048;
constexpr int DE    = 1024;   // d_embed
constexpr int DM    = 4096;   // d_mlp
constexpr int NEXP  = 16;     // experts

constexpr int MT  = 128;      // workgroup M tile
constexpr int NT  = 64;       // workgroup N tile
constexpr int KT  = 32;       // K chunk (one bf16 WMMA K step)
constexpr int AST = 56;       // padded LDS row stride in halves (112B: 16B-aligned,
                              // 28 dwords -> conflict-free b128 frag loads)

// ---------------------------------------------------------------------------
// helpers
// ---------------------------------------------------------------------------

DEV void split_bf16(float f, __bf16& hi, __bf16& lo) {
    unsigned u  = __builtin_bit_cast(unsigned, f);
    unsigned uh = u & 0xFFFF0000u;
    float    fh = __builtin_bit_cast(float, uh);
    float    r  = f - fh;                       // exact residual
    unsigned ur = __builtin_bit_cast(unsigned, r);
    hi = __builtin_bit_cast(__bf16, (unsigned short)(uh >> 16));
    lo = __builtin_bit_cast(__bf16, (unsigned short)(ur >> 16));
}

// ISA 16-bit A 16x32 layout: lanes 0-15 K=0..7 & 16..23, lanes 16-31 K=8..15 & 24..31.
DEV v16bf load_frag(const __bf16* base, int lane) {
    const int c = (lane >> 4) << 3;
    v8bf x0 = *(const v8bf*)(base + c);             // ds_load_b128
    v8bf x1 = *(const v8bf*)(base + 16 + c);        // ds_load_b128
    return __builtin_shufflevector(x0, x1, 0,1,2,3,4,5,6,7,8,9,10,11,12,13,14,15);
}

// Low 32 bits of a flat LDS pointer are the LDS byte offset.
DEV unsigned lds_addr(const void* p) { return (unsigned)(unsigned long long)p; }

// Async DMA: global -> LDS, 16B per lane, tracked by ASYNCcnt (no VGPR round-trip).
DEV void async_b128(unsigned lds_byte, const void* gptr) {
    asm volatile("global_load_async_to_lds_b128 %0, %1, off"
                 :: "v"(lds_byte), "v"(gptr) : "memory");
}
DEV void wait_async6() { asm volatile("s_wait_asynccnt 6" ::: "memory"); }
DEV void wait_async0() { asm volatile("s_wait_asynccnt 0" ::: "memory"); }

// ---------------------------------------------------------------------------
// Pre-split kernels (fast path)
// ---------------------------------------------------------------------------

// Elementwise: f32 row-major -> bf16 hi/lo row-major.
__global__ __launch_bounds__(256)
void split_vec_kernel(const float* __restrict__ src, __bf16* __restrict__ hi,
                      __bf16* __restrict__ lo, size_t n) {
    for (size_t i = (size_t)blockIdx.x * 256 + threadIdx.x; i < n;
         i += (size_t)gridDim.x * 256) {
        split_bf16(src[i], hi[i], lo[i]);
    }
}

// W [NEXP, KDIM, NDIM] f32 -> hi/lo bf16 in chunk-major layout [k][kk/32][n][32].
template <int KDIM, int NDIM>
__global__ __launch_bounds__(256)
void split_w_kernel(const float* __restrict__ W, __bf16* __restrict__ hi,
                    __bf16* __restrict__ lo) {
    __shared__ float tile[KT][256];
    const int n0  = blockIdx.x * 256;
    const int kkc = blockIdx.y;
    const int k   = blockIdx.z;
    const int tid = threadIdx.x;
    const float* Wk = W + (size_t)k * KDIM * NDIM + (size_t)kkc * KT * NDIM + n0;
    for (int i = tid; i < KT * 256; i += 256)
        tile[i >> 8][i & 255] = Wk[(size_t)(i >> 8) * NDIM + (i & 255)];
    __syncthreads();
    // each thread emits one 32-half column (64B contiguous hi, 64B lo)
    const size_t o = ((size_t)(k * (KDIM / KT) + kkc) * NDIM + n0 + tid) * KT;
    __bf16 ch[KT], cl[KT];
#pragma unroll
    for (int h = 0; h < KT; ++h) split_bf16(tile[h][tid], ch[h], cl[h]);
#pragma unroll
    for (int s = 0; s < KT / 8; ++s) {
        *(v8bf*)&hi[o + s * 8] = *(v8bf*)&ch[s * 8];
        *(v8bf*)&lo[o + s * 8] = *(v8bf*)&cl[s * 8];
    }
}

// ---------------------------------------------------------------------------
// Fast-path layer kernel: pre-split operands, async double-buffered staging.
//   inner[b,k,n] = mask[b,k] * (X[b,:] @ W[k,:,:] + bias[k,n])
//   sum[b,n]     = sum_k inner[b,k,n]      (+ optional dup, + optional relu-split)
// ---------------------------------------------------------------------------
template <int KDIM, int NDIM, bool DUAL_OUT, bool WRITE_SPLIT>
__global__ __launch_bounds__(256)
void moe_layer_async_kernel(const __bf16* __restrict__ Xhi,  // [BATCH, KDIM]
                            const __bf16* __restrict__ Xlo,
                            const unsigned char* __restrict__ maskp,
                            const __bf16* __restrict__ Whi,  // [k][kk/32][NDIM][32]
                            const __bf16* __restrict__ Wlo,
                            const float* __restrict__ ball,  // [NEXP, NDIM]
                            float* __restrict__ inner_out,   // [BATCH, NEXP, NDIM]
                            float* __restrict__ sum_out,     // [BATCH, NDIM]
                            float* __restrict__ sum_out2,    // dup (or null)
                            __bf16* __restrict__ Shi,        // relu-split out (or null)
                            __bf16* __restrict__ Slo)
{
    __shared__ __align__(16) __bf16 Ah[2][MT * AST];
    __shared__ __align__(16) __bf16 Al[2][MT * AST];
    __shared__ __align__(16) __bf16 Bh[2][NT * AST];
    __shared__ __align__(16) __bf16 Bl[2][NT * AST];
    __shared__ float maskf[MT * NEXP];

    const int tid  = threadIdx.x;
    const int lane = tid & 31;
    const int wave = tid >> 5;
    const int wr   = wave >> 1;
    const int wc   = wave & 1;
    const int ln   = lane & 15;
    const int lh   = lane >> 4;
    const int row0 = blockIdx.x * MT;
    const int col0 = blockIdx.y * NT;
    constexpr int NCH = KDIM / KT;

    for (int i = tid; i < MT * NEXP; i += 256)
        maskf[i] = maskp[(size_t)(row0 + (i >> 4)) * NEXP + (i & 15)] ? 1.0f : 0.0f;

    // Issue one chunk's async copies (6 instructions per wave: 4 for A, 2 for B).
    auto issue = [&](int kexp, int kkc, int buf) {
        // A tile: 128 rows x 32 halves, hi+lo; 2 chunks of 16B per thread each.
#pragma unroll
        for (int s = 0; s < 2; ++s) {
            const int id = tid + s * 256;
            const int r = id >> 2, c = (id & 3) << 3;
            const size_t go = (size_t)(row0 + r) * KDIM + kkc * KT + c;
            async_b128(lds_addr(&Ah[buf][r * AST + c]), Xhi + go);
            async_b128(lds_addr(&Al[buf][r * AST + c]), Xlo + go);
        }
        // B tile: 64 cols x 32 halves (chunk-major in ws): 1 chunk of 16B per thread.
        {
            const int n = tid >> 2, c = (tid & 3) << 3;
            const size_t go =
                ((size_t)(kexp * NCH + kkc) * NDIM + col0 + n) * KT + c;
            async_b128(lds_addr(&Bh[buf][n * AST + c]), Whi + go);
            async_b128(lds_addr(&Bl[buf][n * AST + c]), Wlo + go);
        }
    };

    const v8f vzero = {0.f, 0.f, 0.f, 0.f, 0.f, 0.f, 0.f, 0.f};
    v8f hacc[2][2];
#pragma unroll
    for (int a = 0; a < 2; ++a)
#pragma unroll
        for (int b = 0; b < 2; ++b) hacc[a][b] = vzero;

    int buf = 0;
    issue(0, 0, 0);

    for (int k = 0; k < NEXP; ++k) {
        v8f acc[2][2];
#pragma unroll
        for (int a = 0; a < 2; ++a)
#pragma unroll
            for (int b = 0; b < 2; ++b) acc[a][b] = vzero;

        for (int kkc = 0; kkc < NCH; ++kkc) {
            // prefetch next chunk into the other buffer
            const bool last = (kkc == NCH - 1) && (k == NEXP - 1);
            if (!last) {
                const int nk = (kkc + 1 < NCH) ? k : k + 1;
                const int nc = (kkc + 1 < NCH) ? kkc + 1 : 0;
                issue(nk, nc, buf ^ 1);
                wait_async6();   // previous chunk resident, next 6 in flight
            } else {
                wait_async0();
            }
            __syncthreads();

#pragma unroll
            for (int tr = 0; tr < 2; ++tr) {
                const __bf16* ar = &Ah[buf][(wr * 32 + tr * 16 + ln) * AST];
                const __bf16* al = &Al[buf][(wr * 32 + tr * 16 + ln) * AST];
                v16bf ah = load_frag(ar, lane);
                v16bf aL = load_frag(al, lane);
#pragma unroll
                for (int tc = 0; tc < 2; ++tc) {
                    const __bf16* br = &Bh[buf][(wc * 32 + tc * 16 + ln) * AST];
                    const __bf16* bl = &Bl[buf][(wc * 32 + tc * 16 + ln) * AST];
                    v16bf bh = load_frag(br, lane);
                    v16bf bL = load_frag(bl, lane);
                    acc[tr][tc] = __builtin_amdgcn_wmma_f32_16x16x32_bf16(
                        false, ah, false, bh, (short)0, acc[tr][tc], false, false);
                    acc[tr][tc] = __builtin_amdgcn_wmma_f32_16x16x32_bf16(
                        false, ah, false, bL, (short)0, acc[tr][tc], false, false);
                    acc[tr][tc] = __builtin_amdgcn_wmma_f32_16x16x32_bf16(
                        false, aL, false, bh, (short)0, acc[tr][tc], false, false);
                }
            }
            __syncthreads();     // all reads of `buf` done before it is refilled
            buf ^= 1;
        }

        // epilogue for expert k
#pragma unroll
        for (int tr = 0; tr < 2; ++tr) {
#pragma unroll
            for (int tc = 0; tc < 2; ++tc) {
                const int colg = col0 + wc * 32 + tc * 16 + ln;
                const float bv = ball[(size_t)k * NDIM + colg];
#pragma unroll
                for (int e = 0; e < 8; ++e) {
                    const int rl = wr * 32 + tr * 16 + (lh << 3) + e;
                    const float mv  = maskf[rl * NEXP + k];
                    const float val = (acc[tr][tc][e] + bv) * mv;
                    inner_out[((size_t)(row0 + rl) * NEXP + k) * NDIM + colg] = val;
                    hacc[tr][tc][e] += val;
                }
            }
        }
    }

#pragma unroll
    for (int tr = 0; tr < 2; ++tr) {
#pragma unroll
        for (int tc = 0; tc < 2; ++tc) {
            const int colg = col0 + wc * 32 + tc * 16 + ln;
#pragma unroll
            for (int e = 0; e < 8; ++e) {
                const int rl = wr * 32 + tr * 16 + (lh << 3) + e;
                const size_t off = (size_t)(row0 + rl) * NDIM + colg;
                const float v = hacc[tr][tc][e];
                sum_out[off] = v;
                if (DUAL_OUT) sum_out2[off] = v;
                if (WRITE_SPLIT) {           // fused relu + split for next layer's A
                    split_bf16(fmaxf(v, 0.0f), Shi[off], Slo[off]);
                }
            }
        }
    }
}

// ---------------------------------------------------------------------------
// Fallback layer kernel (small d_ws): in-kernel f32 -> bf16 split staging.
// ---------------------------------------------------------------------------
template <int KDIM, int NDIM, bool RELU_IN, bool DUAL_OUT>
__global__ __launch_bounds__(256)
void moe_layer_conv_kernel(const float* __restrict__ Xsrc,
                           const unsigned char* __restrict__ maskp,
                           const float* __restrict__ Wall,
                           const float* __restrict__ ball,
                           float* __restrict__ inner_out,
                           float* __restrict__ sum_out,
                           float* __restrict__ sum_out2)
{
    __shared__ __align__(16) __bf16 Ahi[MT * AST];
    __shared__ __align__(16) __bf16 Alo[MT * AST];
    __shared__ __align__(16) __bf16 Bhi[NT * AST];
    __shared__ __align__(16) __bf16 Blo[NT * AST];
    __shared__ float maskf[MT * NEXP];

    const int tid  = threadIdx.x;
    const int lane = tid & 31;
    const int wave = tid >> 5;
    const int wr   = wave >> 1;
    const int wc   = wave & 1;
    const int ln   = lane & 15;
    const int lh   = lane >> 4;
    const int row0 = blockIdx.x * MT;
    const int col0 = blockIdx.y * NT;

    for (int i = tid; i < MT * NEXP; i += 256)
        maskf[i] = maskp[(size_t)(row0 + (i >> 4)) * NEXP + (i & 15)] ? 1.0f : 0.0f;

    const v8f vzero = {0.f, 0.f, 0.f, 0.f, 0.f, 0.f, 0.f, 0.f};
    v8f hacc[2][2];
#pragma unroll
    for (int a = 0; a < 2; ++a)
#pragma unroll
        for (int b = 0; b < 2; ++b) hacc[a][b] = vzero;

    for (int k = 0; k < NEXP; ++k) {
        const float* Wk = Wall + (size_t)k * KDIM * NDIM;
        v8f acc[2][2];
#pragma unroll
        for (int a = 0; a < 2; ++a)
#pragma unroll
            for (int b = 0; b < 2; ++b) acc[a][b] = vzero;

        for (int kk = 0; kk < KDIM; kk += KT) {
            if (kk + KT < KDIM)
                __builtin_prefetch(Wk + (size_t)(kk + KT + (tid >> 3)) * NDIM +
                                   col0 + ((tid & 7) << 3), 0, 1);
            __syncthreads();
            for (int i = tid; i < MT * KT; i += 256) {
                const int r = i >> 5, c = i & (KT - 1);
                float f = Xsrc[(size_t)(row0 + r) * KDIM + kk + c];
                if (RELU_IN) f = fmaxf(f, 0.0f);
                split_bf16(f, Ahi[r * AST + c], Alo[r * AST + c]);
            }
            for (int i = tid; i < KT * NT; i += 256) {
                const int d = i >> 6, n = i & (NT - 1);
                float f = Wk[(size_t)(kk + d) * NDIM + col0 + n];
                split_bf16(f, Bhi[n * AST + d], Blo[n * AST + d]);
            }
            __syncthreads();
#pragma unroll
            for (int tr = 0; tr < 2; ++tr) {
                v16bf ah = load_frag(&Ahi[(wr * 32 + tr * 16 + ln) * AST], lane);
                v16bf aL = load_frag(&Alo[(wr * 32 + tr * 16 + ln) * AST], lane);
#pragma unroll
                for (int tc = 0; tc < 2; ++tc) {
                    v16bf bh = load_frag(&Bhi[(wc * 32 + tc * 16 + ln) * AST], lane);
                    v16bf bL = load_frag(&Blo[(wc * 32 + tc * 16 + ln) * AST], lane);
                    acc[tr][tc] = __builtin_amdgcn_wmma_f32_16x16x32_bf16(
                        false, ah, false, bh, (short)0, acc[tr][tc], false, false);
                    acc[tr][tc] = __builtin_amdgcn_wmma_f32_16x16x32_bf16(
                        false, ah, false, bL, (short)0, acc[tr][tc], false, false);
                    acc[tr][tc] = __builtin_amdgcn_wmma_f32_16x16x32_bf16(
                        false, aL, false, bh, (short)0, acc[tr][tc], false, false);
                }
            }
        }
#pragma unroll
        for (int tr = 0; tr < 2; ++tr) {
#pragma unroll
            for (int tc = 0; tc < 2; ++tc) {
                const int colg = col0 + wc * 32 + tc * 16 + ln;
                const float bv = ball[(size_t)k * NDIM + colg];
#pragma unroll
                for (int e = 0; e < 8; ++e) {
                    const int rl = wr * 32 + tr * 16 + (lh << 3) + e;
                    const float mv  = maskf[rl * NEXP + k];
                    const float val = (acc[tr][tc][e] + bv) * mv;
                    inner_out[((size_t)(row0 + rl) * NEXP + k) * NDIM + colg] = val;
                    hacc[tr][tc][e] += val;
                }
            }
        }
    }
#pragma unroll
    for (int tr = 0; tr < 2; ++tr) {
#pragma unroll
        for (int tc = 0; tc < 2; ++tc) {
            const int colg = col0 + wc * 32 + tc * 16 + ln;
#pragma unroll
            for (int e = 0; e < 8; ++e) {
                const int rl = wr * 32 + tr * 16 + (lh << 3) + e;
                const size_t off = (size_t)(row0 + rl) * NDIM + colg;
                sum_out[off] = hacc[tr][tc][e];
                if (DUAL_OUT) sum_out2[off] = hacc[tr][tc][e];
            }
        }
    }
}

// ---------------------------------------------------------------------------
// host
// ---------------------------------------------------------------------------
extern "C" void kernel_launch(void* const* d_in, const int* in_sizes, int n_in,
                              void* d_out, int out_size, void* d_ws, size_t ws_size,
                              hipStream_t stream) {
    const float*         x    = (const float*)d_in[0];
    const unsigned char* mask = (const unsigned char*)d_in[1];   // jnp bool = 1 byte
    const float*         W1   = (const float*)d_in[2];
    const float*         b1   = (const float*)d_in[3];
    const float*         W2   = (const float*)d_in[4];
    const float*         b2   = (const float*)d_in[5];
    float* out = (float*)d_out;

    // Output tuple layout (flat f32): out, h, out, inner1, inner2
    float* out0   = out;
    float* h      = out0 + (size_t)BATCH * DE;
    float* out2   = h    + (size_t)BATCH * DM;
    float* inner1 = out2 + (size_t)BATCH * DE;
    float* inner2 = inner1 + (size_t)BATCH * NEXP * DM;

    // Scratch layout (bytes) for the pre-split fast path.
    const size_t SZ_X  = (size_t)BATCH * DE * 2;             //  4 MiB each
    const size_t SZ_H  = (size_t)BATCH * DM * 2;             // 16 MiB each
    const size_t SZ_W1 = (size_t)NEXP * DE * DM * 2;         // 128 MiB each
    const size_t SZ_W2 = (size_t)NEXP * DM * DE * 2;         // 128 MiB each
    const size_t NEED  = 2 * (SZ_X + SZ_H + SZ_W1 + SZ_W2);  // 552 MiB

    dim3 blk(256);
    if (ws_size >= NEED) {
        char* p = (char*)d_ws;
        __bf16* xhi  = (__bf16*)p;              p += SZ_X;
        __bf16* xlo  = (__bf16*)p;              p += SZ_X;
        __bf16* hhi  = (__bf16*)p;              p += SZ_H;
        __bf16* hlo  = (__bf16*)p;              p += SZ_H;
        __bf16* w1hi = (__bf16*)p;              p += SZ_W1;
        __bf16* w1lo = (__bf16*)p;              p += SZ_W1;
        __bf16* w2hi = (__bf16*)p;              p += SZ_W2;
        __bf16* w2lo = (__bf16*)p;              p += SZ_W2;

        split_vec_kernel<<<2048, blk, 0, stream>>>(x, xhi, xlo, (size_t)BATCH * DE);
        split_w_kernel<DE, DM><<<dim3(DM / 256, DE / KT, NEXP), blk, 0, stream>>>(W1, w1hi, w1lo);
        split_w_kernel<DM, DE><<<dim3(DE / 256, DM / KT, NEXP), blk, 0, stream>>>(W2, w2hi, w2lo);

        // Layer 1: writes inner1, h, and fused relu+split(h) for layer 2.
        moe_layer_async_kernel<DE, DM, false, true>
            <<<dim3(BATCH / MT, DM / NT), blk, 0, stream>>>(
                xhi, xlo, mask, w1hi, w1lo, b1, inner1, h, nullptr, hhi, hlo);
        // Layer 2: writes inner2 and out (twice).
        moe_layer_async_kernel<DM, DE, true, false>
            <<<dim3(BATCH / MT, DE / NT), blk, 0, stream>>>(
                hhi, hlo, mask, w2hi, w2lo, b2, inner2, out0, out2, nullptr, nullptr);
    } else {
        moe_layer_conv_kernel<DE, DM, false, false>
            <<<dim3(BATCH / MT, DM / NT), blk, 0, stream>>>(x, mask, W1, b1, inner1, h, nullptr);
        moe_layer_conv_kernel<DM, DE, true, true>
            <<<dim3(BATCH / MT, DE / NT), blk, 0, stream>>>(h, mask, W2, b2, inner2, out0, out2);
    }
}
